// QCompass_42021960024798
// MI455X (gfx1250) — matrix operand order repaired and checked by reading, not compiled, not run on hardware
//
#include <hip/hip_runtime.h>

// Problem constants (B=1)
#define LSEQ 2048
#define HDIM 2048
#define RDIM 1024
#define NHEAD 16
#define HR 64    // per-head rank
#define HH 128   // per-head hidden

typedef __bf16 bf16_t;
typedef __attribute__((ext_vector_type(16))) __bf16 v16bf;
typedef __attribute__((ext_vector_type(8)))  __bf16 v8bf;
typedef __attribute__((ext_vector_type(8)))  float  v8f;

// ---------------------------------------------------------------------------
// WMMA fragment loaders (wave32 layouts per CDNA5 ISA 7.12.2).
// Both assume a row-major [rows, K] operand with K contiguous (ldk elements).
// A-matrix 16x32 bf16: lane l16 = row M; half 0 holds K{0..7,16..23},
//                      half 1 holds K{8..15,24..31}.
// B-matrix 32x16 bf16 from Bt[N,K]: lane l16 = col N; half 0 holds K{0..7,16..23},
//                      half 1 holds K{8..15,24..31} (same order as A).
// ---------------------------------------------------------------------------
static __device__ __forceinline__ v16bf load_frag_a(const bf16_t* base, int ldk) {
  const int lane = threadIdx.x & 31;
  const int half = lane >> 4;
  const int l16  = lane & 15;
  const bf16_t* p = base + (size_t)l16 * ldk;
  v8bf lo = *(const v8bf*)(p + half * 8);
  v8bf hi = *(const v8bf*)(p + 16 + half * 8);
  return __builtin_shufflevector(lo, hi, 0,1,2,3,4,5,6,7,8,9,10,11,12,13,14,15);
}

static __device__ __forceinline__ v16bf load_frag_b(const bf16_t* base, int ldk) {
  const int lane = threadIdx.x & 31;
  const int half = lane >> 4;
  const int l16  = lane & 15;
  const bf16_t* p = base + (size_t)l16 * ldk;
  v8bf lo = *(const v8bf*)(p + half * 8);
  v8bf hi = *(const v8bf*)(p + 16 + half * 8);
  return __builtin_shufflevector(lo, hi, 0,1,2,3,4,5,6,7,8,9,10,11,12,13,14,15);
}

static __device__ __forceinline__ v8f wmma_bf16(v16bf a, v16bf b, v8f c) {
  return __builtin_amdgcn_wmma_f32_16x16x32_bf16(false, a, false, b, (short)0, c,
                                                 false, false);
}

// ---------------------------------------------------------------------------
// CDNA5 async global -> LDS copy (16 bytes per lane), tracked by ASYNCcnt.
// vDst operand is the per-lane LDS byte offset (generic shared address
// truncates to addr[31:0] per ISA 10.2).
// ---------------------------------------------------------------------------
static __device__ __forceinline__ void async_copy_b128(const bf16_t* g,
                                                       bf16_t* l) {
  unsigned lofs = (unsigned)(size_t)(void*)l;
  unsigned long long ga = (unsigned long long)(size_t)g;
  asm volatile("global_load_async_to_lds_b128 %0, %1, off"
               :: "v"(lofs), "v"(ga) : "memory");
}
#define WAIT_ASYNC_LE4() asm volatile("s_wait_asynccnt 0x4" ::: "memory")
#define WAIT_ASYNC_LE0() asm volatile("s_wait_asynccnt 0x0" ::: "memory")

// ---------------------------------------------------------------------------
// fp32 -> bf16 cast (grid-stride)
// ---------------------------------------------------------------------------
__global__ void cast_f32_bf16(const float* __restrict__ src,
                              bf16_t* __restrict__ dst, int n) {
  int i = blockIdx.x * blockDim.x + threadIdx.x;
  int stride = gridDim.x * blockDim.x;
  for (; i < n; i += stride) dst[i] = (bf16_t)src[i];
}

// qval = state * action (elementwise, bf16 in/out, f32 math)
__global__ void qval_mul(const bf16_t* __restrict__ s, const bf16_t* __restrict__ a,
                         bf16_t* __restrict__ q, int n) {
  int i = blockIdx.x * blockDim.x + threadIdx.x;
  int stride = gridDim.x * blockDim.x;
  for (; i < n; i += stride) q[i] = (bf16_t)((float)s[i] * (float)a[i]);
}

// ---------------------------------------------------------------------------
// GEMM: C[M,N] = A[M,K] * Bt[N,K]^T   (bf16 inputs, f32 accum via WMMA)
// Block = 256 threads (8 waves), tile 128x128x32.
// Double-buffered LDS fed by global_load_async_to_lds_b128; each thread
// issues 4 async 16B copies per K-tile (2 for A, 2 for B). Async ops
// complete in order, so s_wait_asynccnt<=4 means the older tile is resident
// while the newer tile's copies overlap the WMMAs.
// Wave grid 2x4; wave tile 64(M) x 32(N) = 4x2 WMMA accumulators.
// ---------------------------------------------------------------------------
template<bool STORE_BF16, bool TRANS_OUT, bool ADD_BIAS>
__global__ __launch_bounds__(256)
void gemm_bt(const bf16_t* __restrict__ A, const bf16_t* __restrict__ Bt,
             void* __restrict__ Cout, const float* __restrict__ bias,
             int M, int N, int K) {
  __shared__ __align__(64) bf16_t As[2][128][32];
  __shared__ __align__(64) bf16_t Bs[2][128][32];

  const int tid  = threadIdx.x;
  const int m0   = blockIdx.y * 128;
  const int n0   = blockIdx.x * 128;
  const int wid  = tid >> 5;
  const int wm   = (wid >> 2) * 64;
  const int wn   = (wid & 3) * 32;
  const int lane = tid & 31;
  const int half = lane >> 4;
  const int l16  = lane & 15;

  const int lrow = tid >> 1;         // 0..127 : tile row this thread stages
  const int lseg = (tid & 1) * 16;   // 16-element segment within K-tile

  // Per-thread global staging pointers (advance by 32 per K-tile)
  const bf16_t* gA = A  + (size_t)(m0 + lrow) * K + lseg;
  const bf16_t* gB = Bt + (size_t)(n0 + lrow) * K + lseg;

  v8f acc[4][2];
  #pragma unroll
  for (int i = 0; i < 4; ++i) {
    #pragma unroll
    for (int j = 0; j < 2; ++j) {
      v8f z = {0.f,0.f,0.f,0.f,0.f,0.f,0.f,0.f};
      acc[i][j] = z;
    }
  }

  const int ktiles = K >> 5;

  // Prologue: stage tile 0 into buffer 0
  async_copy_b128(gA,     &As[0][lrow][lseg]);
  async_copy_b128(gA + 8, &As[0][lrow][lseg + 8]);
  async_copy_b128(gB,     &Bs[0][lrow][lseg]);
  async_copy_b128(gB + 8, &Bs[0][lrow][lseg + 8]);

  for (int kt = 0; kt < ktiles; ++kt) {
    const int cur = kt & 1;
    if (kt + 1 < ktiles) {
      const bf16_t* nA = gA + (kt + 1) * 32;
      const bf16_t* nB = gB + (kt + 1) * 32;
      async_copy_b128(nA,     &As[cur ^ 1][lrow][lseg]);
      async_copy_b128(nA + 8, &As[cur ^ 1][lrow][lseg + 8]);
      async_copy_b128(nB,     &Bs[cur ^ 1][lrow][lseg]);
      async_copy_b128(nB + 8, &Bs[cur ^ 1][lrow][lseg + 8]);
      WAIT_ASYNC_LE4();               // tile kt resident; kt+1 in flight
    } else {
      WAIT_ASYNC_LE0();               // last tile resident
    }
    __syncthreads();                  // cross-wave visibility of LDS tile

    v16bf af[4], bfv[2];
    #pragma unroll
    for (int i = 0; i < 4; ++i) af[i]  = load_frag_a(&As[cur][wm + i * 16][0], 32);
    #pragma unroll
    for (int j = 0; j < 2; ++j) bfv[j] = load_frag_b(&Bs[cur][wn + j * 16][0], 32);
    #pragma unroll
    for (int i = 0; i < 4; ++i) {
      #pragma unroll
      for (int j = 0; j < 2; ++j) acc[i][j] = wmma_bf16(af[i], bfv[j], acc[i][j]);
    }
    __syncthreads();                  // buffer reusable for tile kt+2
  }

  // Epilogue: C/D layout -> element (m,n) lives in lane (m>>3)*16+n, vgpr m&7
  #pragma unroll
  for (int i = 0; i < 4; ++i) {
    #pragma unroll
    for (int j = 0; j < 2; ++j) {
      #pragma unroll
      for (int v = 0; v < 8; ++v) {
        const int row = m0 + wm + i * 16 + half * 8 + v;
        const int col = n0 + wn + j * 16 + l16;
        float val = acc[i][j][v];
        if (ADD_BIAS) val += bias[col];
        if (STORE_BF16) {
          bf16_t* C = (bf16_t*)Cout;
          if (TRANS_OUT) C[(size_t)col * M + row] = (bf16_t)val;
          else           C[(size_t)row * N + col] = (bf16_t)val;
        } else {
          ((float*)Cout)[(size_t)row * N + col] = val;
        }
      }
    }
  }
}

// ---------------------------------------------------------------------------
// Fused causal routing attention (flash-style, online softmax).
// One wave per (head, 16-query tile). K per-head rank = 64 => 2 WMMA k-steps.
// contentT is [H, L] (pre-transposed) so the P*V step is also a [N,K] operand.
// ---------------------------------------------------------------------------
__global__ __launch_bounds__(32)
void qroute_attn(const bf16_t* __restrict__ state_b,
                 const bf16_t* __restrict__ action_b,
                 const bf16_t* __restrict__ contentT,
                 bf16_t* __restrict__ gathered) {
  __shared__ __align__(64) bf16_t Pt[16][32];   // P tile for C->A relayout

  const int qt    = blockIdx.x;        // 0..L/16-1
  const int h     = blockIdx.y;        // 0..NH-1
  const int qbase = qt * 16;
  const int lane  = threadIdx.x & 31;
  const int half  = lane >> 4;
  const int l16   = lane & 15;

  // Q fragments held in registers for the whole kernel (K = 64 -> 2 steps)
  const bf16_t* qptr = state_b + (size_t)qbase * RDIM + h * HR;
  v16bf aq[2];
  #pragma unroll
  for (int ks = 0; ks < 2; ++ks) aq[ks] = load_frag_a(qptr + ks * 32, RDIM);

  v8f o[8];
  float mrun[8], lrun[8];
  #pragma unroll
  for (int j = 0; j < 8; ++j) { v8f z = {0.f,0.f,0.f,0.f,0.f,0.f,0.f,0.f}; o[j] = z; }
  #pragma unroll
  for (int v = 0; v < 8; ++v) { mrun[v] = -3.0e38f; lrun[v] = 0.f; }

  const int ntiles = (qbase + 16 + 31) >> 5;    // key tiles of 32 (causal)
  for (int t = 0; t < ntiles; ++t) {
    const int kb = t * 32;

    // S = Qh * Kh^T : two 16-wide key subtiles
    v8f s0 = {0.f,0.f,0.f,0.f,0.f,0.f,0.f,0.f};
    v8f s1 = s0;
    const bf16_t* kptr = action_b + (size_t)kb * RDIM + h * HR;
    #pragma unroll
    for (int ks = 0; ks < 2; ++ks) {
      v16bf b0 = load_frag_b(kptr, RDIM);
      v16bf b1 = load_frag_b(kptr + (size_t)16 * RDIM, RDIM);
      s0 = wmma_bf16(aq[ks], b0, s0);
      s1 = wmma_bf16(aq[ks], b1, s1);
      kptr += 32;
    }

    // Online softmax (scale 1/sqrt(64) = 0.125, causal mask k<=q)
    float p0[8], p1[8], alpha[8];
    #pragma unroll
    for (int v = 0; v < 8; ++v) {
      const int qrow = qbase + half * 8 + v;
      float e0 = (kb + l16      <= qrow) ? s0[v] * 0.125f : -3.0e38f;
      float e1 = (kb + 16 + l16 <= qrow) ? s1[v] * 0.125f : -3.0e38f;
      float rm = fmaxf(e0, e1);
      #pragma unroll
      for (int mk = 1; mk < 16; mk <<= 1) rm = fmaxf(rm, __shfl_xor(rm, mk, 32));
      const float mnew = fmaxf(mrun[v], rm);
      const float a  = __expf(mrun[v] - mnew);
      const float q0 = __expf(e0 - mnew);
      const float q1 = __expf(e1 - mnew);
      float rs = q0 + q1;
      #pragma unroll
      for (int mk = 1; mk < 16; mk <<= 1) rs += __shfl_xor(rs, mk, 32);
      lrun[v] = lrun[v] * a + rs;
      mrun[v] = mnew;
      alpha[v] = a; p0[v] = q0; p1[v] = q1;
    }

    // Rescale accumulated O by per-row alpha (row ownership matches C layout)
    #pragma unroll
    for (int j = 0; j < 8; ++j) {
      #pragma unroll
      for (int v = 0; v < 8; ++v) o[j][v] *= alpha[v];
    }

    // C-layout P -> LDS -> A-layout fragment
    __syncthreads();
    #pragma unroll
    for (int v = 0; v < 8; ++v) {
      Pt[half * 8 + v][l16]      = (bf16_t)p0[v];
      Pt[half * 8 + v][16 + l16] = (bf16_t)p1[v];
    }
    __syncthreads();
    v16bf pa = load_frag_a(&Pt[0][0], 32);

    // O += P * Vh  (Vh from transposed content: [Hh rows, L cols])
    #pragma unroll
    for (int j = 0; j < 8; ++j) {
      const bf16_t* vp = contentT + (size_t)(h * HH + j * 16) * LSEQ + kb;
      v16bf bv = load_frag_b(vp, LSEQ);
      o[j] = wmma_bf16(pa, bv, o[j]);
    }
  }

  // Normalize and store gathered[L,H] (bf16, A-operand for the final GEMM)
  #pragma unroll
  for (int v = 0; v < 8; ++v) {
    const float inv = 1.f / lrun[v];
    const int row = qbase + half * 8 + v;
    #pragma unroll
    for (int j = 0; j < 8; ++j) {
      gathered[(size_t)row * HDIM + h * HH + j * 16 + l16] = (bf16_t)(o[j][v] * inv);
    }
  }
}

// ---------------------------------------------------------------------------
// Launcher
// ---------------------------------------------------------------------------
extern "C" void kernel_launch(void* const* d_in, const int* in_sizes, int n_in,
                              void* d_out, int out_size, void* d_ws, size_t ws_size,
                              hipStream_t stream) {
  const float* x  = (const float*)d_in[0];
  const float* Ws = (const float*)d_in[1];
  const float* Wa = (const float*)d_in[2];
  const float* Wc = (const float*)d_in[3];
  const float* Wo = (const float*)d_in[4];
  const float* bo = (const float*)d_in[5];
  float* out = (float*)d_out;

  // Workspace layout (bf16), total ~56 MB
  bf16_t* xb  = (bf16_t*)d_ws;
  bf16_t* Wsb = xb  + (size_t)LSEQ * HDIM;
  bf16_t* Wab = Wsb + (size_t)RDIM * HDIM;
  bf16_t* Wcb = Wab + (size_t)RDIM * HDIM;
  bf16_t* Wob = Wcb + (size_t)HDIM * RDIM;
  bf16_t* stb = Wob + (size_t)HDIM * HDIM;   // state  [L,R]
  bf16_t* acb = stb + (size_t)LSEQ * RDIM;   // action [L,R]
  bf16_t* qvb = acb + (size_t)LSEQ * RDIM;   // qval   [L,R]
  bf16_t* ctT = qvb + (size_t)LSEQ * RDIM;   // content^T [H,L]
  bf16_t* gat = ctT + (size_t)HDIM * LSEQ;   // gathered [L,H]

  const int CB = 256;
  cast_f32_bf16<<<1024, CB, 0, stream>>>(x,  xb,  LSEQ * HDIM);
  cast_f32_bf16<<<512,  CB, 0, stream>>>(Ws, Wsb, RDIM * HDIM);
  cast_f32_bf16<<<512,  CB, 0, stream>>>(Wa, Wab, RDIM * HDIM);
  cast_f32_bf16<<<512,  CB, 0, stream>>>(Wc, Wcb, HDIM * RDIM);
  cast_f32_bf16<<<1024, CB, 0, stream>>>(Wo, Wob, HDIM * HDIM);

  dim3 blk(256);
  // state / action projections
  gemm_bt<true,  false, false><<<dim3(RDIM/128, LSEQ/128), blk, 0, stream>>>(
      xb, Wsb, stb, nullptr, LSEQ, RDIM, HDIM);
  gemm_bt<true,  false, false><<<dim3(RDIM/128, LSEQ/128), blk, 0, stream>>>(
      xb, Wab, acb, nullptr, LSEQ, RDIM, HDIM);
  // elementwise q-value
  qval_mul<<<1024, CB, 0, stream>>>(stb, acb, qvb, LSEQ * RDIM);
  // content projection, stored transposed [H,L]
  gemm_bt<true,  true,  false><<<dim3(HDIM/128, LSEQ/128), blk, 0, stream>>>(
      qvb, Wcb, ctT, nullptr, LSEQ, HDIM, RDIM);
  // fused causal routing attention
  qroute_attn<<<dim3(LSEQ/16, NHEAD), dim3(32), 0, stream>>>(stb, acb, ctT, gat);
  // output projection + bias, f32 out
  gemm_bt<false, false, true ><<<dim3(HDIM/128, LSEQ/128), blk, 0, stream>>>(
      gat, Wob, (void*)out, bo, LSEQ, HDIM, HDIM);
}
